// MyModel_34909494181927
// MI455X (gfx1250) — matrix operand order repaired and checked by reading, not compiled, not run on hardware
//
#include <hip/hip_runtime.h>

typedef __attribute__((ext_vector_type(16))) _Float16 v16h;
typedef __attribute__((ext_vector_type(8)))  float    v8f;

#define IMG_H 360
#define IMG_W 640
#define NCH   196     // = 4 * 7 * 7
#define OUTS  7
#define GC    4
#define CT    13      // ceil(196/16) channel tiles (208 cols padded)
#define RMAX  66      // max box region extent after rounding
#define PADR  68      // region + 2-px halo
#define PMMAX (((RMAX * RMAX + 15) / 16) * 16)   // padded pixel-mask table

// Fused 3x3 SAME conv (3->196) + bias + ReLU + PS-ROI-pool + bin mean.
// One workgroup per box. Implicit GEMM, transposed vs usual conv:
//   A [16 pix x 32 K]  = im2col of a pixel tile (built per tile from LDS)
//   B [32 K x 16 ch]   = weights (+bias row K=27, A supplies 1.0), hoisted in VGPRs
//   C [16 pix x 16 ch] -> lane owns ONE channel, 8 pixel rows => branchless
//                         register-accumulated PS-ROI pooling.
__global__ __launch_bounds__(256)
void psroi_conv_fused(const float* __restrict__ mv,
                      const float* __restrict__ boxes,
                      const float* __restrict__ scl,
                      const float* __restrict__ cw,
                      const float* __restrict__ cb,
                      float* __restrict__ out)
{
    __shared__ _Float16 sIn[3][PADR][PADR];            // haloed f16 input region
    __shared__ __align__(32) _Float16 sB[CT][32][16];  // WMMA B fragments (weights)
    __shared__ float sBin[CT * 16];                    // per-channel bin sums
    __shared__ __align__(16) unsigned short sPM[PMMAX];// per-pixel maskY|maskX<<8
    __shared__ unsigned char sMY[PADR];                // bin mask per region row
    __shared__ unsigned char sMX[PADR];                // bin mask per region col
    __shared__ int sHS[OUTS], sHE[OUTS], sWS[OUTS], sWE[OUTS];

    const int k    = blockIdx.x;
    const int tid  = threadIdx.x;
    const int lane = tid & 31;
    const int wv   = tid >> 5;

    // ---- box geometry (uniform per block) ----
    const float scale = scl[0];
    const float* bx = boxes + k * 5;
    const int   bi = (int)bx[0];
    const float x1 = rintf(bx[1] * scale);
    const float y1 = rintf(bx[2] * scale);
    const float x2 = rintf(bx[3] * scale);
    const float y2 = rintf(bx[4] * scale);
    const float rw = fmaxf(x2 - x1, 0.1f);
    const float rh = fmaxf(y2 - y1, 0.1f);
    const float bh = rh * (1.0f / OUTS);
    const float bw = rw * (1.0f / OUTS);

    if (tid < OUTS) {
        const int p = tid;
        sHS[p] = min(max((int)floorf((float)p * bh + y1), 0), IMG_H);
        sHE[p] = min(max((int)ceilf(((float)p + 1.0f) * bh + y1), 0), IMG_H);
        sWS[p] = min(max((int)floorf((float)p * bw + x1), 0), IMG_W);
        sWE[p] = min(max((int)ceilf(((float)p + 1.0f) * bw + x1), 0), IMG_W);
    }

    const int hs0 = min(max((int)floorf(y1), 0), IMG_H);
    const int he6 = min(max((int)ceilf(rh + y1), 0), IMG_H);
    const int ws0 = min(max((int)floorf(x1), 0), IMG_W);
    const int we6 = min(max((int)ceilf(rw + x1), 0), IMG_W);
    int RH = min(he6 - hs0, RMAX); if (RH < 0) RH = 0;
    int RW = min(we6 - ws0, RMAX); if (RW < 0) RW = 0;

    // ---- stage haloed input region to LDS as f16 (SAME padding -> zeros) ----
    const int HR = RH + 2, WR = RW + 2;
    for (int idx = tid; idx < 3 * HR * WR; idx += 256) {
        const int c = idx / (HR * WR);
        const int r = (idx / WR) % HR;
        const int q = idx % WR;
        const int gy = hs0 - 1 + r;
        const int gx = ws0 - 1 + q;
        float v = 0.0f;
        if (gy >= 0 && gy < IMG_H && gx >= 0 && gx < IMG_W)
            v = mv[(((size_t)bi * 3 + c) * IMG_H + gy) * IMG_W + gx];
        sIn[c][r][q] = (_Float16)v;
    }

    // ---- pack weights (+bias row K=27) into B fragments, contiguous k order as in A ----
    // B is K x N: lane L holds column n = L%16; element e -> K = (L/16)*16 + e.
    for (int idx = tid; idx < CT * 32 * 16; idx += 256) {
        const int ct = idx >> 9;
        const int ln = (idx >> 4) & 31;
        const int e  = idx & 15;
        const int ch = ct * 16 + (ln & 15);
        const int kk = (ln < 16 ? 0 : 16) + e;
        _Float16 v = (_Float16)0.0f;
        if (ch < NCH) {
            if (kk < 27)       v = (_Float16)cw[ch * 27 + kk];
            else if (kk == 27) v = (_Float16)cb[ch];          // bias row
        }
        sB[ct][ln][e] = v;
    }
    for (int idx = tid; idx < CT * 16; idx += 256) sBin[idx] = 0.0f;
    __syncthreads();

    // ---- per-row / per-col bin membership masks (7 bits each) ----
    for (int idx = tid; idx < RH; idx += 256) {
        const int gy = hs0 + idx;
        unsigned m = 0;
        for (int ph = 0; ph < OUTS; ++ph)
            if (gy >= sHS[ph] && gy < sHE[ph]) m |= 1u << ph;
        sMY[idx] = (unsigned char)m;
    }
    for (int idx = tid; idx < RW; idx += 256) {
        const int gx = ws0 + idx;
        unsigned m = 0;
        for (int pw = 0; pw < OUTS; ++pw)
            if (gx >= sWS[pw] && gx < sWE[pw]) m |= 1u << pw;
        sMX[idx] = (unsigned char)m;
    }
    __syncthreads();

    // ---- linear per-pixel combined mask table (zero-padded tail) ----
    const int Np     = RH * RW;
    const int ntiles = (Np + 15) >> 4;
    for (int idx = tid; idx < ntiles * 16; idx += 256) {
        unsigned short v = 0;
        if (idx < Np) {
            const int py = idx / RW, px = idx % RW;
            v = (unsigned short)((unsigned)sMY[py] | ((unsigned)sMX[px] << 8));
        }
        sPM[idx] = v;
    }

    // ---- hoist all 13 weight B-fragments into VGPRs ----
    v16h bfr[CT];
#pragma unroll
    for (int ct = 0; ct < CT; ++ct)
        bfr[ct] = *(const v16h*)(&sB[ct][lane][0]);

    // per-lane channel bin masks: lm = (1<<ph) | (256<<pw); invalid -> bit 16
    const int ncol = lane & 15;
    unsigned lm[CT];
#pragma unroll
    for (int ct = 0; ct < CT; ++ct) {
        const int ch = ct * 16 + ncol;
        unsigned m = 0x10000u;                      // never matches 16-bit pm
        if (ch < NCH) {
            const int rem = ch % 49;
            m = (1u << (rem / 7)) | (256u << (rem % 7));
        }
        lm[ct] = m;
    }

    // per-lane im2col LDS byte offsets (A fragment: M=pixel, K split by half)
    const int khalf = lane >> 4;
    int aoff[16];
#pragma unroll
    for (int e = 0; e < 16; ++e) {
        const int K = (khalf ? 16 : 0) + e;
        int off = 0;
        if (K < 27) {
            const int c = K / 9, r = K % 9;
            off = (c * PADR * PADR + (r / 3) * PADR + (r % 3)) * 2;
        }
        aoff[e] = off;
    }

    float acc[CT];
#pragma unroll
    for (int ct = 0; ct < CT; ++ct) acc[ct] = 0.0f;

    __syncthreads();   // sPM ready

    const int voff = (lane < 16) ? 0 : 8;
    for (int nt = wv; nt < ntiles; nt += 8) {          // uniform per wave
        const int  p    = nt * 16 + ncol;              // this lane's A-row pixel
        const int  pp   = (p < Np) ? p : 0;
        const int  py   = pp / RW;
        const int  px   = pp % RW;

        // A fragment: 16 unconditional LDS u16 loads off per-pixel base
        const char* base = (const char*)(&sIn[0][py][px]);
        v16h afr = {};
#pragma unroll
        for (int e = 0; e < 16; ++e)
            afr[e] = *(const _Float16*)(base + aoff[e]);
        afr[11] = khalf ? (_Float16)1.0f : afr[11];    // K=27 bias row = 1.0

        // masks for this lane's 8 C-row pixels: one aligned b128
        const uint4 pmv = *(const uint4*)(&sPM[nt * 16 + voff]);
        unsigned pmx[8];
        pmx[0] = pmv.x & 0xFFFFu;  pmx[1] = pmv.x >> 16;
        pmx[2] = pmv.y & 0xFFFFu;  pmx[3] = pmv.y >> 16;
        pmx[4] = pmv.z & 0xFFFFu;  pmx[5] = pmv.z >> 16;
        pmx[6] = pmv.w & 0xFFFFu;  pmx[7] = pmv.w >> 16;

#pragma unroll
        for (int ct = 0; ct < CT; ++ct) {
            v8f cfr = {};
            cfr = __builtin_amdgcn_wmma_f32_16x16x32_f16(
                false, afr, false, bfr[ct], (short)0, cfr, false, false);
#pragma unroll
            for (int v = 0; v < 8; ++v) {
                // branchless: in-bin ? relu(conv+bias) : 0, register accumulate
                const float t = ((pmx[v] & lm[ct]) == lm[ct]) ? cfr[v] : 0.0f;
                acc[ct] += fmaxf(t, 0.0f);
            }
        }
    }

    // ---- flush register accumulators (once per lane) ----
#pragma unroll
    for (int ct = 0; ct < CT; ++ct)
        atomicAdd(&sBin[ct * 16 + ncol], acc[ct]);     // ds_add_f32
    __syncthreads();

    // ---- final: mean over 7x7 of (bin sum / analytic area) ----
    if (tid < GC) {
        float s = 0.0f;
        for (int i = 0; i < 49; ++i) {
            const int ph = i / 7, pw = i % 7;
            const int area = (sHE[ph] - sHS[ph]) * (sWE[pw] - sWS[pw]);
            if (area > 0) s += sBin[tid * 49 + i] / (float)area;
        }
        out[k * GC + tid] = s * (1.0f / 49.0f);
    }
}

extern "C" void kernel_launch(void* const* d_in, const int* in_sizes, int n_in,
                              void* d_out, int out_size, void* d_ws, size_t ws_size,
                              hipStream_t stream) {
    (void)n_in; (void)out_size; (void)d_ws; (void)ws_size;
    const float* mv    = (const float*)d_in[0];   // [2,3,360,640]
    const float* boxes = (const float*)d_in[1];   // [128,5]
    const float* scl   = (const float*)d_in[2];   // [1,2]
    const float* cw    = (const float*)d_in[3];   // [196,3,3,3]
    const float* cb    = (const float*)d_in[4];   // [196]
    float* out = (float*)d_out;                   // [128,4]
    const int K = in_sizes[1] / 5;
    psroi_conv_fused<<<K, 256, 0, stream>>>(mv, boxes, scl, cw, cb, out);
}
